// HarmonicSynthesizer_56118042689682
// MI455X (gfx1250) — matrix operand order repaired and checked by reading, not compiled, not run on hardware
//
#include <hip/hip_runtime.h>
#include <hip/hip_bf16.h>
#include <math.h>

typedef _Float16 v16h __attribute__((ext_vector_type(16)));
typedef _Float16 v8h  __attribute__((ext_vector_type(8)));
typedef float    v8f  __attribute__((ext_vector_type(8)));

union HV { v16h v16; v8h v8[2]; };

#define PI_F 3.14159265358979323846f
#define SR_F 24000.0f

// ---------------------------------------------------------------------------
// Weight prep: f32 -> f16 (pw3 padded from 100 to 128 output rows with zeros)
// ---------------------------------------------------------------------------
__global__ void prep_weights_kernel(const float* __restrict__ w1,
                                    const float* __restrict__ w2,
                                    const float* __restrict__ w3,
                                    _Float16* __restrict__ W1h,   // [256,128]
                                    _Float16* __restrict__ W2h,   // [256,256]
                                    _Float16* __restrict__ W3h,   // [128,256] padded
                                    int n1, int n2, int K3, int Hout, int Mpad) {
  int idx = blockIdx.x * blockDim.x + threadIdx.x;
  if (idx < n1) W1h[idx] = (_Float16)w1[idx];
  if (idx < n2) W2h[idx] = (_Float16)w2[idx];
  int n3 = Mpad * K3;
  if (idx < n3) {
    int r = idx / K3;
    float v = (r < Hout) ? w3[idx] : 0.0f;
    W3h[idx] = (_Float16)v;
  }
}

// ---------------------------------------------------------------------------
// Depthwise conv 1: condition [B,C,T] f32 -> x1T [N=B*T, C] f16
// ---------------------------------------------------------------------------
__global__ void dw1_kernel(const float* __restrict__ x,
                           const float* __restrict__ w,
                           const float* __restrict__ bias,
                           _Float16* __restrict__ y,
                           int B, int C, int T) {
  int idx = blockIdx.x * blockDim.x + threadIdx.x;
  int c = idx % C;
  int n = idx / C;
  if (n >= B * T) return;
  int b = n / T, t = n % T;
  const float* xr = x + ((size_t)(b * C + c)) * T + t;
  float xl = (t > 0)     ? xr[-1] : 0.0f;
  float xm = xr[0];
  float xrt = (t < T - 1) ? xr[1] : 0.0f;
  float v = w[c * 3 + 0] * xl + w[c * 3 + 1] * xm + w[c * 3 + 2] * xrt + bias[c];
  y[(size_t)n * C + c] = (_Float16)v;
}

// ---------------------------------------------------------------------------
// Depthwise conv 2: x [N,C] f16 -> y [N,C] f16 (zero-pad at batch t edges)
// ---------------------------------------------------------------------------
__global__ void dw2_kernel(const _Float16* __restrict__ x,
                           const float* __restrict__ w,
                           const float* __restrict__ bias,
                           _Float16* __restrict__ y,
                           int B, int C, int T) {
  int idx = blockIdx.x * blockDim.x + threadIdx.x;
  int c = idx % C;
  int n = idx / C;
  if (n >= B * T) return;
  int t = n % T;
  float xl = (t > 0)     ? (float)x[(size_t)(n - 1) * C + c] : 0.0f;
  float xm = (float)x[(size_t)n * C + c];
  float xr = (t < T - 1) ? (float)x[(size_t)(n + 1) * C + c] : 0.0f;
  float v = w[c * 3 + 0] * xl + w[c * 3 + 1] * xm + w[c * 3 + 2] * xr + bias[c];
  y[(size_t)n * C + c] = (_Float16)v;
}

// ---------------------------------------------------------------------------
// WMMA GEMM: D[M,N] = act(W[M,KC] @ X^T + bias). X stored [N,KC] f16.
// One 16x16 n-tile per block; 8 waves * MT m-tiles each cover M.
// EPI 0: leaky-relu -> f16 out [N,M].  EPI 1: softplus -> f32 ampsT[N,AST], h<Hout.
// ---------------------------------------------------------------------------
template <int KC, int MT, int EPI>
__global__ void gemm_wmma_kernel(const _Float16* __restrict__ W,
                                 const _Float16* __restrict__ X,
                                 const float* __restrict__ bias,
                                 _Float16* __restrict__ outH,
                                 float* __restrict__ outF,
                                 int M, int AST, int Hout) {
  const int lane = threadIdx.x & 31;
  const int wave = threadIdx.x >> 5;
  const int row  = lane & 15;   // A row within tile == C/D column within tile
  const int half = lane >> 4;
  const int n = blockIdx.x * 16 + row;

  const _Float16* __restrict__ Xrow = X + (size_t)n * KC;

  v8f acc[MT] = {};
#pragma unroll
  for (int kt = 0; kt < KC / 32; ++kt) {
    const int kb = kt * 32;
    HV bf;
    bf.v8[0] = *(const v8h*)(Xrow + kb + half * 16);
    bf.v8[1] = *(const v8h*)(Xrow + kb + half * 16 + 8);
#pragma unroll
    for (int i = 0; i < MT; ++i) {
      const int mt = wave * MT + i;
      const _Float16* __restrict__ Wrow =
          W + (size_t)(mt * 16 + row) * KC + kb + half * 16;
      HV af;
      af.v8[0] = *(const v8h*)(Wrow);
      af.v8[1] = *(const v8h*)(Wrow + 8);
      acc[i] = __builtin_amdgcn_wmma_f32_16x16x32_f16(
          false, af.v16, false, bf.v16, (short)0, acc[i], false, false);
    }
  }

#pragma unroll
  for (int i = 0; i < MT; ++i) {
    const int mt = wave * MT + i;
    const int mb = mt * 16 + half * 8;   // 8 contiguous output channels
    if constexpr (EPI == 0) {
      v8h o;
#pragma unroll
      for (int v = 0; v < 8; ++v) {
        float f = acc[i][v] + bias[mb + v];
        f = (f >= 0.0f) ? f : 0.1f * f;
        o[v] = (_Float16)f;
      }
      *(v8h*)(outH + (size_t)n * M + mb) = o;
    } else {
#pragma unroll
      for (int v = 0; v < 8; ++v) {
        int h = mb + v;
        if (h < Hout) {
          float f = acc[i][v] + bias[h];
          float sp = (f > 20.0f) ? f : log1pf(__expf(f));
          outF[(size_t)n * AST + h] = sp;
        }
      }
    }
  }
}

// ---------------------------------------------------------------------------
// Phase: per-batch cumsum of 2*pi*f0_up/SR over L samples. One block per batch.
// ---------------------------------------------------------------------------
__device__ __forceinline__ float f0_interp(const float* f0s, int l, int T, float TdivL) {
  float src = (l + 0.5f) * TdivL - 0.5f;
  src = fminf(fmaxf(src, 0.0f), (float)(T - 1));
  int lo = (int)floorf(src);
  int hi = min(lo + 1, T - 1);
  float w = src - (float)lo;
  return f0s[lo] * (1.0f - w) + f0s[hi] * w;
}

__global__ void phase_kernel(const float* __restrict__ f0,
                             float* __restrict__ phase,
                             int T, int L) {
  __shared__ float f0s[512];
  __shared__ float psum[256];
  const int b = blockIdx.x;
  const int tid = threadIdx.x;
  for (int i = tid; i < T; i += blockDim.x) f0s[i] = f0[(size_t)b * T + i];
  __syncthreads();

  const float TdivL = (float)T / (float)L;
  const float k = 2.0f * PI_F / SR_F;
  const int chunk = (L + blockDim.x - 1) / blockDim.x;
  const int l0 = tid * chunk;
  const int l1 = min(l0 + chunk, L);

  float sum = 0.0f;
  for (int l = l0; l < l1; ++l) sum += k * f0_interp(f0s, l, T, TdivL);
  psum[tid] = sum;
  __syncthreads();
  for (int off = 1; off < 256; off <<= 1) {
    float add = (tid >= off) ? psum[tid - off] : 0.0f;
    __syncthreads();
    psum[tid] += add;
    __syncthreads();
  }
  float run = psum[tid] - sum;   // exclusive prefix
  float* __restrict__ ph = phase + (size_t)b * L;
  for (int l = l0; l < l1; ++l) {
    run += k * f0_interp(f0s, l, T, TdivL);
    ph[l] = run;                 // inclusive cumsum
  }
}

// ---------------------------------------------------------------------------
// Oscillator bank: block = (frame j, batch b), 240 samples. Harmonic recurrence.
// ---------------------------------------------------------------------------
__global__ void synth_kernel(const float* __restrict__ ampsT,  // [B*T, AST]
                             const float* __restrict__ phase,  // [B, L]
                             const float* __restrict__ f0,     // [B, T]
                             float* __restrict__ out,          // [B, L]
                             int T, int L, int SPF, int AST) {
  constexpr int H = 100;
  __shared__ float S[3 * H];
  const int j = blockIdx.x;
  const int b = blockIdx.y;

  const int fr0 = max(j - 1, 0);
  const int fr2 = min(j + 1, T - 1);
  const int frames[3] = {fr0, j, fr2};
  for (int i = threadIdx.x; i < 3 * H; i += blockDim.x) {
    int fr = i / H, h = i % H;
    S[fr * H + h] = ampsT[(size_t)(b * T + frames[fr]) * AST + h];
  }
  __syncthreads();

  const int u = threadIdx.x;
  if (u >= SPF) return;
  const int l = j * SPF + u;

  float src = (l + 0.5f) * ((float)T / (float)L) - 0.5f;
  src = fminf(fmaxf(src, 0.0f), (float)(T - 1));
  int lo = (int)floorf(src);
  int hi = min(lo + 1, T - 1);
  float w = src - (float)lo;

  const float* __restrict__ f0r = f0 + (size_t)b * T;
  float f0u = f0r[lo] * (1.0f - w) + f0r[hi] * w;

  float p = phase[(size_t)b * L + l];
  float s1 = __sinf(p), c1 = __cosf(p);
  float s = s1, c = c1;

  int lo_rel = min(max(lo - (j - 1), 0), 2);
  int hi_rel = min(max(hi - (j - 1), 0), 2);
  const float* __restrict__ Alo = S + lo_rel * H;
  const float* __restrict__ Ahi = S + hi_rel * H;

  const float nyq = 0.5f * SR_F;
  int hmax = min(H, (int)(nyq / f0u));
  while (hmax > 0 && (float)hmax * f0u >= nyq) --hmax;   // strict h*f0 < SR/2

  float acc = 0.0f;
  for (int h = 1; h <= hmax; ++h) {
    float a0 = Alo[h - 1], a1 = Ahi[h - 1];
    acc = fmaf(a0 + w * (a1 - a0), s, acc);
    float ns = fmaf(s, c1, c * s1);           // sin((h+1)p)
    c = fmaf(c, c1, -s * s1);                 // cos((h+1)p)
    s = ns;
  }
  out[(size_t)b * L + l] = acc;
}

// ---------------------------------------------------------------------------
extern "C" void kernel_launch(void* const* d_in, const int* in_sizes, int n_in,
                              void* d_out, int out_size, void* d_ws, size_t ws_size,
                              hipStream_t stream) {
  const float* f0     = (const float*)d_in[0];
  const float* cond   = (const float*)d_in[1];
  const float* dw1_w  = (const float*)d_in[2];
  const float* dw1_b  = (const float*)d_in[3];
  const float* pw1_w  = (const float*)d_in[4];
  const float* pw1_b  = (const float*)d_in[5];
  const float* dw2_w  = (const float*)d_in[6];
  const float* dw2_b  = (const float*)d_in[7];
  const float* pw2_w  = (const float*)d_in[8];
  const float* pw2_b  = (const float*)d_in[9];
  const float* pw3_w  = (const float*)d_in[10];
  const float* pw3_b  = (const float*)d_in[11];
  float* out = (float*)d_out;

  const int B = 8;
  const int T = in_sizes[0] / B;        // 500
  const int L = out_size / B;           // 120000
  const int SPF = L / T;                // 240
  const int N = B * T;                  // 4000
  const int C1 = 128, C2 = 256, H = 100, MP = 128, AST = 112;

  // carve workspace (256B aligned blocks)
  char* p = (char*)d_ws;
  auto carve = [&](size_t bytes) {
    void* r = (void*)p;
    p += (bytes + 255) & ~(size_t)255;
    return r;
  };
  _Float16* W1h   = (_Float16*)carve((size_t)C2 * C1 * 2);
  _Float16* W2h   = (_Float16*)carve((size_t)C2 * C2 * 2);
  _Float16* W3h   = (_Float16*)carve((size_t)MP * C2 * 2);
  _Float16* x1T   = (_Float16*)carve((size_t)N * C1 * 2);
  _Float16* x2T   = (_Float16*)carve((size_t)N * C2 * 2);
  _Float16* x3T   = (_Float16*)carve((size_t)N * C2 * 2);
  _Float16* x4T   = (_Float16*)carve((size_t)N * C2 * 2);
  float*    ampsT = (float*)   carve((size_t)N * AST * 4);
  float*    phase = (float*)   carve((size_t)B * L * 4);

  // 1. weights -> f16 (pw3 zero-padded to 128 rows)
  prep_weights_kernel<<<(C2 * C2 + 255) / 256, 256, 0, stream>>>(
      pw1_w, pw2_w, pw3_w, W1h, W2h, W3h, C2 * C1, C2 * C2, C2, H, MP);

  // 2. depthwise 1
  dw1_kernel<<<(N * C1 + 255) / 256, 256, 0, stream>>>(
      cond, dw1_w, dw1_b, x1T, B, C1, T);

  // 3. pointwise 1 (WMMA): [256x128] @ [128,N], lrelu -> f16
  gemm_wmma_kernel<128, 2, 0><<<N / 16, 256, 0, stream>>>(
      W1h, x1T, pw1_b, x2T, nullptr, C2, 0, 0);

  // 4. depthwise 2
  dw2_kernel<<<(N * C2 + 255) / 256, 256, 0, stream>>>(
      x2T, dw2_w, dw2_b, x3T, B, C2, T);

  // 5. pointwise 2 (WMMA): [256x256] @ [256,N], lrelu -> f16
  gemm_wmma_kernel<256, 2, 0><<<N / 16, 256, 0, stream>>>(
      W2h, x3T, pw2_b, x4T, nullptr, C2, 0, 0);

  // 6. pointwise 3 (WMMA): [128(pad)x256] @ [256,N], softplus -> f32 ampsT (h<100)
  gemm_wmma_kernel<256, 1, 1><<<N / 16, 256, 0, stream>>>(
      W3h, x4T, pw3_b, nullptr, ampsT, MP, AST, H);

  // 7. phase cumsum (independent of 1-6; stream-ordered anyway)
  phase_kernel<<<B, 256, 0, stream>>>(f0, phase, T, L);

  // 8. oscillator bank
  synth_kernel<<<dim3(T, B), 256, 0, stream>>>(
      ampsT, phase, f0, out, T, L, SPF, AST);
}
